// Net_34325378630270
// MI455X (gfx1250) — matrix operand order repaired and checked by reading, not compiled, not run on hardware
//
#include <hip/hip_runtime.h>

// ---------------------------------------------------------------------------
// Types for CDNA5 WMMA (wave32, 16x16x32 bf16 -> f32 accumulate)
// ---------------------------------------------------------------------------
typedef __bf16 bf16;
typedef __attribute__((ext_vector_type(8)))  __bf16 v8bf;
typedef __attribute__((ext_vector_type(16))) __bf16 v16bf;
typedef __attribute__((ext_vector_type(4)))  float  v4f;
typedef __attribute__((ext_vector_type(8)))  float  v8f;

union BF16x16 { v16bf v; v8bf h[2]; };

static constexpr int NB_   = 64;    // batch
static constexpr int NL    = 512;   // seq len
static constexpr int NC    = 256;   // block channels
static constexpr int ND    = 512;   // concat channels (2 timeframes)
static constexpr int NFF   = 1024;  // ff channels
static constexpr int EMBP  = 160;   // EMB_DIM 144 padded to multiple of 32

// ---------------------------------------------------------------------------
// NT-GEMM (+K-tap conv) main loop, cloned for boundary-masked vs interior
// waves so interior waves carry no exec-mask manipulation at all.
//   acc[mi][t] += A[k][m0+mi*16..][.] . B[n0+t*16+shift(k)..][.]
// ---------------------------------------------------------------------------
template<int KTAPS, bool CHECK>
__device__ __forceinline__
void ntg_mainloop(const bf16* __restrict__ Ab, const bf16* __restrict__ Bb,
                  int M, int N, int Kd, int m0, int n0, int n, int hi,
                  v8f acc[2][4])
{
    v16bf bzero;
#pragma unroll
    for (int i = 0; i < 16; ++i) bzero[i] = static_cast<__bf16>(0.0f);

#pragma unroll
    for (int k = 0; k < KTAPS; ++k) {
        const int shift = (KTAPS == 1) ? 0 : (k - KTAPS / 2);
        const bf16* Ak = Ab + (size_t)k * M * Kd;
        for (int cb = 0; cb < Kd; cb += 32) {
            const bf16* ap0 = Ak + (size_t)(m0 + n) * Kd + cb + hi * 8;
            const bf16* ap1 = ap0 + (size_t)16 * Kd;
            BF16x16 a0, a1;
            a0.h[0] = *(const v8bf*)ap0;
            a0.h[1] = *(const v8bf*)(ap0 + 16);
            a1.h[0] = *(const v8bf*)ap1;
            a1.h[1] = *(const v8bf*)(ap1 + 16);
#pragma unroll
            for (int t = 0; t < 4; ++t) {
                int row = n0 + t * 16 + n + shift;
                BF16x16 bfrag;
                if (!CHECK || (row >= 0 && row < N)) {
                    const bf16* bp = Bb + (size_t)row * Kd + cb + hi * 8;
                    bfrag.h[0] = *(const v8bf*)bp;
                    bfrag.h[1] = *(const v8bf*)(bp + 16);
                } else {
                    bfrag.v = bzero;
                }
                acc[0][t] = __builtin_amdgcn_wmma_f32_16x16x32_bf16(
                    false, a0.v, false, bfrag.v, (short)0, acc[0][t], false, false);
                acc[1][t] = __builtin_amdgcn_wmma_f32_16x16x32_bf16(
                    false, a1.v, false, bfrag.v, (short)0, acc[1][t], false, false);
            }
        }
    }
}

// ---------------------------------------------------------------------------
// Generic NT-GEMM (+K-tap conv) kernel:
//   D[M][N] = scale * sum_{k,K} A[k][M][Kd] . B[N+shift(k)][Kd]  (+bias, +PE, relu)
// stored to Y[N][M] (contiguous). One wave per 32(M) x 64(N) strip.
// A-fragment (bf16 16x32): lane<16 -> row m0+lane, K {0..7,16..23}; lane>=16 K {8..15,24..31}
// B-fragment (bf16 32x16): lane<16 -> col n, K {0..7,16..23}; lane>=16 K {8..15,24..31}
// D-tile f32: lane n holds (M=hi*8+r, N=n) -> 8 contiguous channels at fixed row
// ---------------------------------------------------------------------------
template<int KTAPS, bool HAS_BIAS, bool ADD_PE, bool RELU, bool OUT_F32, bool OUT_BF16>
__global__ __launch_bounds__(32)
void ntgemm_kernel(const bf16* __restrict__ A, const bf16* __restrict__ Bm,
                   const float* __restrict__ bias, float scale,
                   float* __restrict__ Yf, bf16* __restrict__ Yb,
                   int M, int N, int Kd, long aBatch, long bBatch)
{
    const int lane = threadIdx.x;
    const int n    = lane & 15;
    const int hi   = lane >> 4;
    const int m0   = blockIdx.y * 32;           // 2 M-tiles per wave
    const int n0   = blockIdx.x * 64;           // 4 N-tiles per wave
    const int b    = blockIdx.z;

    const bf16* Ab = A  + (size_t)aBatch * b;
    const bf16* Bb = Bm + (size_t)bBatch * b;

    v8f acc[2][4];
#pragma unroll
    for (int mi = 0; mi < 2; ++mi)
#pragma unroll
        for (int t = 0; t < 4; ++t)
#pragma unroll
            for (int r = 0; r < 8; ++r) acc[mi][t][r] = 0.0f;

    const int half = KTAPS / 2;
    const bool interior = (KTAPS == 1) || ((n0 >= half) && (n0 + 64 + half <= N));
    if (interior)
        ntg_mainloop<KTAPS, false>(Ab, Bb, M, N, Kd, m0, n0, n, hi, acc);
    else
        ntg_mainloop<KTAPS, true>(Ab, Bb, M, N, Kd, m0, n0, n, hi, acc);

#pragma unroll
    for (int mi = 0; mi < 2; ++mi) {
        const int mbase = m0 + mi * 16 + hi * 8;
#pragma unroll
        for (int t = 0; t < 4; ++t) {
            int row = n0 + t * 16 + n;          // output N index
            float vals[8];
#pragma unroll
            for (int r = 0; r < 8; ++r) {
                float x = acc[mi][t][r] * scale;
                int c = mbase + r;
                if (HAS_BIAS) x += bias[c];
                if (ADD_PE) {
                    float ce  = (float)(c & ~1);
                    float div = __expf(ce * (-9.210340371976184f / 256.0f)); // -ln(1e4)/C
                    float ang = div * (float)row;
                    x += (c & 1) ? __cosf(ang) : __sinf(ang);
                }
                if (RELU) x = fmaxf(x, 0.0f);
                vals[r] = x;
            }
            size_t off = (size_t)b * N * M + (size_t)row * M + mbase;
            if (OUT_F32) {
                v4f o0, o1;
#pragma unroll
                for (int r = 0; r < 4; ++r) { o0[r] = vals[r]; o1[r] = vals[4 + r]; }
                *(v4f*)(Yf + off)     = o0;
                *(v4f*)(Yf + off + 4) = o1;
            }
            if (OUT_BF16) {
                v8bf ob;
#pragma unroll
                for (int r = 0; r < 8; ++r) ob[r] = static_cast<__bf16>(vals[r]);
                *(v8bf*)(Yb + off) = ob;
            }
        }
    }
}

template<int KTAPS, bool HAS_BIAS, bool ADD_PE, bool RELU, bool OUT_F32, bool OUT_BF16>
static void launch_ntgemm(const bf16* A, const bf16* Bm, const float* bias, float scale,
                          float* Yf, bf16* Yb, int M, int N, int Kd,
                          long aBatch, long bBatch, hipStream_t s)
{
    dim3 grid(N / 64, M / 32, NB_);
    ntgemm_kernel<KTAPS, HAS_BIAS, ADD_PE, RELU, OUT_F32, OUT_BF16>
        <<<grid, 32, 0, s>>>(A, Bm, bias, scale, Yf, Yb, M, N, Kd, aBatch, bBatch);
}

// ---------------------------------------------------------------------------
// Weight repack: f32 [Cout][Cin][K] -> bf16 [K][Cout][Cinp] (zero-padded Cin)
// ---------------------------------------------------------------------------
__global__ void convert_w_kernel(const float* __restrict__ src, bf16* __restrict__ dst,
                                 int Cout, int Cin, int Cinp, int K)
{
    int idx = blockIdx.x * blockDim.x + threadIdx.x;
    int total = K * Cout * Cinp;
    if (idx >= total) return;
    int ci = idx % Cinp;
    int r  = idx / Cinp;
    int co = r % Cout;
    int k  = r / Cout;
    dst[idx] = (ci < Cin) ? static_cast<__bf16>(src[((size_t)co * Cin + ci) * K + k])
                          : static_cast<__bf16>(0.0f);
}

// ---------------------------------------------------------------------------
// Periodic embedding + linear + relu + cat embedding -> xemb [B][L][160] bf16
// ---------------------------------------------------------------------------
__global__ __launch_bounds__(128)
void embed_kernel(const float* __restrict__ num, const int* __restrict__ cat,
                  const float* __restrict__ per,     // [4][16]
                  const float* __restrict__ lin_w,   // [4][32][32]
                  const float* __restrict__ lin_b,   // [4][32]
                  const float* __restrict__ cat_emb, // [18][16]
                  bf16* __restrict__ xemb)
{
    int idx = blockIdx.x * blockDim.x + threadIdx.x;
    if (idx >= NB_ * NL) return;
    const float* x = num + (size_t)idx * 4;
    bf16* out = xemb + (size_t)idx * EMBP;

    for (int f = 0; f < 4; ++f) {
        float xs[16], xc[16];
        float xv = x[f];
#pragma unroll
        for (int c = 0; c < 16; ++c) {
            float ang = xv * per[f * 16 + c] * 6.283185307179586f;
            xs[c] = __sinf(ang);
            xc[c] = __cosf(ang);
        }
        for (int d = 0; d < 32; ++d) {
            const float* w = lin_w + (size_t)(f * 32) * 32 + d;
            float s = lin_b[f * 32 + d];
#pragma unroll
            for (int c = 0; c < 16; ++c) {
                s += xs[c] * w[c * 32];
                s += xc[c] * w[(16 + c) * 32];
            }
            out[f * 32 + d] = static_cast<__bf16>(fmaxf(s, 0.0f));
        }
    }
    int ci = cat[idx];
#pragma unroll
    for (int j = 0; j < 16; ++j) out[128 + j] = static_cast<__bf16>(cat_emb[ci * 16 + j]);
#pragma unroll
    for (int j = 144; j < 160; ++j) out[j] = static_cast<__bf16>(0.0f);
}

// ---------------------------------------------------------------------------
// LDS-tiled transpose: qkv [B][L][768] -> qT/kT [B][512][512] rows dOff..dOff+255
// ---------------------------------------------------------------------------
__global__ void transpose_qk_kernel(const bf16* __restrict__ qkv,
                                    bf16* __restrict__ qT, bf16* __restrict__ kT, int dOff)
{
    __shared__ bf16 tile[32][33];
    int tY   = blockIdx.y;            // 0..15 : 0-7 -> q, 8-15 -> k
    bool isK = tY >= 8;
    int c0   = (tY & 7) * 32;
    int l0   = blockIdx.x * 32;
    int b    = blockIdx.z;
    const bf16* src = qkv + (size_t)b * NL * 768 + (isK ? 256 : 0);
    for (int i = threadIdx.y; i < 32; i += 8)
        tile[i][threadIdx.x] = src[(size_t)(l0 + i) * 768 + c0 + threadIdx.x];
    __syncthreads();
    bf16* dst = (isK ? kT : qT) + (size_t)b * ND * NL + (size_t)(dOff + c0) * NL + l0;
    for (int i = threadIdx.y; i < 32; i += 8)
        dst[(size_t)i * NL + threadIdx.x] = tile[threadIdx.x][i];
}

// v gather: vc[b][l][e] from the two qkv buffers (channels 512..767)
__global__ void gather_v_kernel(const bf16* __restrict__ q1, const bf16* __restrict__ q15,
                                bf16* __restrict__ vc)
{
    int idx = blockIdx.x * blockDim.x + threadIdx.x;
    if (idx >= NB_ * NL * ND) return;
    int e  = idx & 511;
    int bl = idx >> 9;  // b*L + l
    bf16 v = (e < 256) ? q1[(size_t)bl * 768 + 512 + e]
                       : q15[(size_t)bl * 768 + 512 + (e - 256)];
    vc[idx] = v;
}

// ---------------------------------------------------------------------------
// Softmax over d (axis=1) : logits stored [b][e][d] (rows contiguous),
// output scattered into attn [b][d][e] bf16 for the next NT-GEMM.
// ---------------------------------------------------------------------------
__global__ __launch_bounds__(128)
void softmax_kernel(const float* __restrict__ logits, bf16* __restrict__ attn)
{
    __shared__ float redm[4];
    __shared__ float reds[4];
    int e = blockIdx.x, b = blockIdx.y;
    int t = threadIdx.x;
    const float* row = logits + ((size_t)b * ND + e) * ND;
    float v[4];
    float m = -3.4e38f;
#pragma unroll
    for (int i = 0; i < 4; ++i) { v[i] = row[t + i * 128]; m = fmaxf(m, v[i]); }
#pragma unroll
    for (int o = 16; o > 0; o >>= 1) m = fmaxf(m, __shfl_xor(m, o, 32));
    if ((t & 31) == 0) redm[t >> 5] = m;
    __syncthreads();
    m = fmaxf(fmaxf(redm[0], redm[1]), fmaxf(redm[2], redm[3]));
    float s = 0.0f;
#pragma unroll
    for (int i = 0; i < 4; ++i) { v[i] = __expf(v[i] - m); s += v[i]; }
#pragma unroll
    for (int o = 16; o > 0; o >>= 1) s += __shfl_xor(s, o, 32);
    if ((t & 31) == 0) reds[t >> 5] = s;
    __syncthreads();
    float inv = 1.0f / (reds[0] + reds[1] + reds[2] + reds[3]);
#pragma unroll
    for (int i = 0; i < 4; ++i) {
        int d = t + i * 128;
        attn[((size_t)b * ND + d) * ND + e] = static_cast<__bf16>(v[i] * inv);
    }
}

// ---------------------------------------------------------------------------
// residual add + LayerNorm over 256 channels; one wave per (b,l)
// ---------------------------------------------------------------------------
__global__ __launch_bounds__(32)
void add_ln_kernel(const float* __restrict__ base, const float* __restrict__ add,
                   int addOff, int addStride,
                   const float* __restrict__ g, const float* __restrict__ bta,
                   float* __restrict__ out32, bf16* __restrict__ out16)
{
    int bl   = blockIdx.x;
    int lane = threadIdx.x;
    const float* bp = base + (size_t)bl * NC;
    const float* ap = add + (size_t)bl * addStride + addOff;
    float x[8];
    float s = 0.0f;
#pragma unroll
    for (int i = 0; i < 8; ++i) { x[i] = bp[i * 32 + lane] + ap[i * 32 + lane]; s += x[i]; }
#pragma unroll
    for (int o = 16; o > 0; o >>= 1) s += __shfl_xor(s, o, 32);
    float mu = s * (1.0f / NC);
    float vs = 0.0f;
#pragma unroll
    for (int i = 0; i < 8; ++i) { float d = x[i] - mu; vs += d * d; }
#pragma unroll
    for (int o = 16; o > 0; o >>= 1) vs += __shfl_xor(vs, o, 32);
    float inv = rsqrtf(vs * (1.0f / NC) + 1e-5f);
#pragma unroll
    for (int i = 0; i < 8; ++i) {
        int c = i * 32 + lane;
        float y = (x[i] - mu) * inv * g[c] + bta[c];
        out32[(size_t)bl * NC + c] = y;
        out16[(size_t)bl * NC + c] = static_cast<__bf16>(y);
    }
}

// ---------------------------------------------------------------------------
// Head MLP: [256 last-step feats | 16 sym emb] -> 512 -> 256 -> 4
// ---------------------------------------------------------------------------
__global__ __launch_bounds__(256)
void head_kernel(const float* __restrict__ y1, const float* __restrict__ sym_emb,
                 const int* __restrict__ sym_idx,
                 const float* __restrict__ w0, const float* __restrict__ b0,
                 const float* __restrict__ w1, const float* __restrict__ b1,
                 const float* __restrict__ w2, const float* __restrict__ b2,
                 float* __restrict__ out)
{
    __shared__ float z0[272];
    __shared__ float z1[512];
    __shared__ float z2[256];
    int b = blockIdx.x, t = threadIdx.x;
    if (t < 256) z0[t] = y1[((size_t)b * NL + (NL - 1)) * NC + t];
    if (t < 16)  z0[256 + t] = sym_emb[sym_idx[b] * 16 + t];
    __syncthreads();
    for (int j = t; j < 512; j += 256) {
        float s = b0[j];
        for (int i = 0; i < 272; ++i) s += z0[i] * w0[(size_t)i * 512 + j];
        z1[j] = fmaxf(s, 0.0f);
    }
    __syncthreads();
    if (t < 256) {
        float s = b1[t];
        for (int i = 0; i < 512; ++i) s += z1[i] * w1[(size_t)i * 256 + t];
        z2[t] = fmaxf(s, 0.0f);
    }
    __syncthreads();
    if (t < 4) {
        float s = b2[t];
        for (int i = 0; i < 256; ++i) s += z2[i] * w2[(size_t)i * 4 + t];
        out[b * 4 + t] = s;
    }
}

// ---------------------------------------------------------------------------
// Host launcher
// ---------------------------------------------------------------------------
extern "C" void kernel_launch(void* const* d_in, const int* in_sizes, int n_in,
                              void* d_out, int out_size, void* d_ws, size_t ws_size,
                              hipStream_t stream)
{
    (void)in_sizes; (void)n_in; (void)out_size; (void)ws_size;
    // ---- input pointers (jax tree flatten order: sorted dict keys, lists ordered)
    const float* head_b[3] = {(const float*)d_in[0], (const float*)d_in[2], (const float*)d_in[4]};
    const float* head_w[3] = {(const float*)d_in[1], (const float*)d_in[3], (const float*)d_in[5]};
    const float *ln_b1[4], *ln_b2[4], *ln_g1[4], *ln_g2[4];
    for (int i = 0; i < 4; ++i) {
        ln_b1[i] = (const float*)d_in[6 + 4 * i];
        ln_b2[i] = (const float*)d_in[7 + 4 * i];
        ln_g1[i] = (const float*)d_in[8 + 4 * i];
        ln_g2[i] = (const float*)d_in[9 + 4 * i];
    }
    const float *ff1_b[2][4], *ff1_w[2][4], *ff2_b[2][4], *ff2_w[2][4], *qkv_b[2][4], *qkv_w[2][4];
    const float *cat_emb[2], *emb_b[2], *emb_w[2], *lin_b[2], *lin_w[2], *per[2];
    int tfbase[2] = {22, 52};
    for (int tf = 0; tf < 2; ++tf) {
        int base = tfbase[tf];
        for (int k = 0; k < 4; ++k) {
            ff1_b[tf][k] = (const float*)d_in[base + 6 * k + 0];
            ff1_w[tf][k] = (const float*)d_in[base + 6 * k + 1];
            ff2_b[tf][k] = (const float*)d_in[base + 6 * k + 2];
            ff2_w[tf][k] = (const float*)d_in[base + 6 * k + 3];
            qkv_b[tf][k] = (const float*)d_in[base + 6 * k + 4];
            qkv_w[tf][k] = (const float*)d_in[base + 6 * k + 5];
        }
        cat_emb[tf] = (const float*)d_in[base + 24];
        emb_b[tf]   = (const float*)d_in[base + 25];
        emb_w[tf]   = (const float*)d_in[base + 26];
        lin_b[tf]   = (const float*)d_in[base + 27];
        lin_w[tf]   = (const float*)d_in[base + 28];
        per[tf]     = (const float*)d_in[base + 29];
    }
    const float* sym_emb = (const float*)d_in[82];
    const float* num_in[2] = {(const float*)d_in[83], (const float*)d_in[85]};
    const int*   cat_in[2] = {(const int*)d_in[84],   (const int*)d_in[86]};
    const int*   sym_idx   = (const int*)d_in[87];

    // ---- workspace carve
    char* ws = (char*)d_ws;
    size_t cur = 0;
    auto alloc = [&](size_t bytes) -> void* {
        void* p = ws + cur;
        cur = (cur + bytes + 255) & ~(size_t)255;
        return p;
    };
    bf16* wEmb[2];  for (int t = 0; t < 2; ++t) wEmb[t] = (bf16*)alloc((size_t)5 * NC * EMBP * 2);
    bf16 *wQkv[2][4], *wFf1[2][4], *wFf2[2][4];
    for (int t = 0; t < 2; ++t)
        for (int k = 0; k < 4; ++k) {
            wQkv[t][k] = (bf16*)alloc((size_t)5 * 768 * NC * 2);
            wFf1[t][k] = (bf16*)alloc((size_t)5 * NFF * NC * 2);
            wFf2[t][k] = (bf16*)alloc((size_t)5 * NC * NFF * 2);
        }
    bf16*  xemb[2]; for (int t = 0; t < 2; ++t) xemb[t] = (bf16*)alloc((size_t)NB_ * NL * EMBP * 2);
    float* y32[2];  for (int t = 0; t < 2; ++t) y32[t]  = (float*)alloc((size_t)NB_ * NL * NC * 4);
    bf16*  y16[2];  for (int t = 0; t < 2; ++t) y16[t]  = (bf16*)alloc((size_t)NB_ * NL * NC * 2);
    float* xln32[2];for (int t = 0; t < 2; ++t) xln32[t]= (float*)alloc((size_t)NB_ * NL * NC * 4);
    bf16*  xln16[2];for (int t = 0; t < 2; ++t) xln16[t]= (bf16*)alloc((size_t)NB_ * NL * NC * 2);
    bf16*  qkv16[2];for (int t = 0; t < 2; ++t) qkv16[t]= (bf16*)alloc((size_t)NB_ * NL * 768 * 2);
    bf16*  qT     = (bf16*)alloc((size_t)NB_ * ND * NL * 2);
    bf16*  kT     = (bf16*)alloc((size_t)NB_ * ND * NL * 2);
    bf16*  vc     = (bf16*)alloc((size_t)NB_ * NL * ND * 2);
    float* logits = (float*)alloc((size_t)NB_ * ND * ND * 4);
    bf16*  attn   = (bf16*)alloc((size_t)NB_ * ND * ND * 2);
    float* aout   = (float*)alloc((size_t)NB_ * NL * ND * 4);
    bf16*  ff1buf = (bf16*)alloc((size_t)NB_ * NL * NFF * 2);
    float* ff2buf = (float*)alloc((size_t)NB_ * NL * NC * 4);

    auto cdiv = [](int a, int b) { return (a + b - 1) / b; };

    // ---- weight repack (f32 [Cout][Cin][K] -> bf16 [K][Cout][Cinp])
    for (int t = 0; t < 2; ++t) {
        int tot = 5 * NC * EMBP;
        convert_w_kernel<<<cdiv(tot, 256), 256, 0, stream>>>(emb_w[t], wEmb[t], NC, 144, EMBP, 5);
        for (int k = 0; k < 4; ++k) {
            tot = 5 * 768 * NC;
            convert_w_kernel<<<cdiv(tot, 256), 256, 0, stream>>>(qkv_w[t][k], wQkv[t][k], 768, NC, NC, 5);
            tot = 5 * NFF * NC;
            convert_w_kernel<<<cdiv(tot, 256), 256, 0, stream>>>(ff1_w[t][k], wFf1[t][k], NFF, NC, NC, 5);
            tot = 5 * NC * NFF;
            convert_w_kernel<<<cdiv(tot, 256), 256, 0, stream>>>(ff2_w[t][k], wFf2[t][k], NC, NFF, NFF, 5);
        }
    }

    // ---- embedding: periodic features + cat emb, then conv (WMMA) + bias + PE
    for (int t = 0; t < 2; ++t) {
        embed_kernel<<<cdiv(NB_ * NL, 128), 128, 0, stream>>>(
            num_in[t], cat_in[t], per[t], lin_w[t], lin_b[t], cat_emb[t], xemb[t]);
        launch_ntgemm<5, true, true, false, true, true>(
            wEmb[t], xemb[t], emb_b[t], 1.0f, y32[t], y16[t],
            NC, NL, EMBP, 0, (long)NL * EMBP, stream);
    }

    // ---- BlockNet stack
    const float rsqL = 0.04419417382415922f; // 1/sqrt(512)
    for (int blk = 0; blk < 4; ++blk) {
        for (int t = 0; t < 2; ++t)
            launch_ntgemm<5, true, false, false, false, true>(
                wQkv[t][blk], y16[t], qkv_b[t][blk], 1.0f, nullptr, qkv16[t],
                768, NL, NC, 0, (long)NL * NC, stream);
        for (int t = 0; t < 2; ++t)
            transpose_qk_kernel<<<dim3(NL / 32, 16, NB_), dim3(32, 8), 0, stream>>>(
                qkv16[t], qT, kT, t * 256);
        launch_ntgemm<1, false, false, false, true, false>(
            qT, kT, nullptr, rsqL, logits, nullptr,
            ND, ND, NL, (long)ND * NL, (long)ND * NL, stream);
        softmax_kernel<<<dim3(ND, NB_), 128, 0, stream>>>(logits, attn);
        gather_v_kernel<<<cdiv(NB_ * NL * ND, 256), 256, 0, stream>>>(qkv16[0], qkv16[1], vc);
        launch_ntgemm<1, false, false, false, true, false>(
            attn, vc, nullptr, 1.0f, aout, nullptr,
            ND, NL, ND, (long)ND * ND, (long)NL * ND, stream);
        for (int t = 0; t < 2; ++t)
            add_ln_kernel<<<NB_ * NL, 32, 0, stream>>>(
                y32[t], aout, t * 256, ND, ln_g1[blk], ln_b1[blk], xln32[t], xln16[t]);
        for (int t = 0; t < 2; ++t) {
            launch_ntgemm<5, true, false, true, false, true>(
                wFf1[t][blk], xln16[t], ff1_b[t][blk], 1.0f, nullptr, ff1buf,
                NFF, NL, NC, 0, (long)NL * NC, stream);
            launch_ntgemm<5, true, false, false, true, false>(
                wFf2[t][blk], ff1buf, ff2_b[t][blk], 1.0f, ff2buf, nullptr,
                NC, NL, NFF, 0, (long)NL * NFF, stream);
            add_ln_kernel<<<NB_ * NL, 32, 0, stream>>>(
                xln32[t], ff2buf, 0, NC, ln_g2[blk], ln_b2[blk], y32[t], y16[t]);
        }
    }

    // ---- head
    head_kernel<<<NB_, 256, 0, stream>>>(
        y32[0], sym_emb, sym_idx,
        head_w[0], head_b[0], head_w[1], head_b[1], head_w[2], head_b[2],
        (float*)d_out);
}
